// GOKGModel_83270825935425
// MI455X (gfx1250) — matrix operand order repaired and checked by reading, not compiled, not run on hardware
//
#include <hip/hip_runtime.h>

typedef __attribute__((ext_vector_type(16))) _Float16 v16h;
typedef __attribute__((ext_vector_type(8)))  _Float16 v8h;
typedef __attribute__((ext_vector_type(4)))  _Float16 v4h;
typedef __attribute__((ext_vector_type(2)))  _Float16 v2h;
typedef __attribute__((ext_vector_type(8)))  float    v8f;

#define WMMA_F32_F16(a, b, c) \
  __builtin_amdgcn_wmma_f32_16x16x32_f16(false, (a), false, (b), (short)0, (c), false, false)

#define CAT16(lo, hi) \
  __builtin_shufflevector((lo), (hi), 0, 1, 2, 3, 4, 5, 6, 7, 8, 9, 10, 11, 12, 13, 14, 15)

// ---- WMMA fragment loads as 2x ds_load_b128 each ---------------------------
// A (16x32 MxK, row-major f16 tile in LDS): lane's 16 values are two
// contiguous 8xf16 runs: k = ks*32+base+[0..7] and ks*32+base+8+[0..7].
__device__ __forceinline__ v16h frag_a(const _Float16* As, int lane, int Kld, int ks) {
  const int row  = lane & 15;
  const int base = (lane >> 4) << 4;
  const _Float16* p = As + row * Kld + (ks << 5) + base;
  const v8h lo = *(const v8h*)(p);
  const v8h hi = *(const v8h*)(p + 8);
  return CAT16(lo, hi);
}
// B (32x16 KxN) from pre-swizzled LDS: each lane's 16 values are contiguous.
// Layout: Wswz[((c*S + ks)*32 + lane)*16 + i], value = W[k][n],
//         k = ks*32 + (lane>>4)*16 + i,  n = c*16 + (lane&15).
__device__ __forceinline__ v16h frag_b(const _Float16* Wswz, int lane, int S, int c, int ks) {
  const _Float16* p = Wswz + ((((c * S + ks) << 5) + lane) << 4);
  const v8h lo = *(const v8h*)(p);
  const v8h hi = *(const v8h*)(p + 8);
  return CAT16(lo, hi);
}

// Fill LDS with weights pre-swizzled into per-lane fragment order.
// Two consecutive i share (lane,c,ks): pack 2 f32 -> 2 f16, store b32.
template <int K, int N>
__device__ __forceinline__ void fill_w_swz(_Float16* Wswz, const float* __restrict__ W,
                                           int tid, int nthr) {
  constexpr int S = K / 32;  // power of two
  for (int idx2 = tid; idx2 < (K * N) / 2; idx2 += nthr) {
    const int idx  = idx2 << 1;
    const int i    = idx & 15;      // even
    const int lane = (idx >> 4) & 31;
    const int cks  = idx >> 9;      // c*S + ks
    const int ks   = cks & (S - 1);
    const int c    = cks / S;
    const int k = (ks << 5) + ((lane >> 4) << 4) + i;
    const int n = (c << 4) + (lane & 15);
    v2h h;
    h[0] = (_Float16)W[k * N + n];
    h[1] = (_Float16)W[(k + 1) * N + n];
    *(v2h*)(Wswz + idx) = h;
  }
}

// Fill a 16-row fp32 activation tile into LDS as f16: float4 -> v4h.
template <int K>
__device__ __forceinline__ void fill_a_tile(_Float16* As, const float* __restrict__ A,
                                            long long row0, int tid, int nthr) {
  for (int i4 = tid; i4 < (16 * K) / 4; i4 += nthr) {
    const int i = i4 << 2;
    const int r = i / K, c = i & (K - 1);
    const float4 f = *(const float4*)(A + (row0 + r) * K + c);
    v4h h;
    h[0] = (_Float16)f.x; h[1] = (_Float16)f.y;
    h[2] = (_Float16)f.z; h[3] = (_Float16)f.w;
    *(v4h*)(As + i) = h;
  }
}

// ---- out = relu(A[MxK] @ W[KxN] + bias), one 16-row tile / block -----------
template <int K, int N>
__global__ void __launch_bounds__((N / 16) * 32)
gemm_bias_relu(const float* __restrict__ A, const float* __restrict__ W,
               const float* __restrict__ bias, float* __restrict__ out) {
  __shared__ alignas(16) _Float16 Ws[K * N];
  __shared__ alignas(16) _Float16 As[16 * K];
  constexpr int S = K / 32;
  const int tid  = threadIdx.x;
  const int nthr = (N / 16) * 32;
  const long long row0 = (long long)blockIdx.x * 16;

  fill_w_swz<K, N>(Ws, W, tid, nthr);
  fill_a_tile<K>(As, A, row0, tid, nthr);
  __syncthreads();

  const int wave = tid >> 5, lane = tid & 31;
  v8f acc = {};
#pragma unroll
  for (int ks = 0; ks < S; ++ks)
    acc = WMMA_F32_F16(frag_a(As, lane, K, ks), frag_b(Ws, lane, S, wave, ks), acc);

  const int col = wave * 16 + (lane & 15);
  const float bv = bias[col];
#pragma unroll
  for (int r = 0; r < 8; ++r) {
    const int rr = r + ((lane >> 4) << 3);
    float v = acc[r] + bv;
    out[(row0 + rr) * N + col] = v > 0.f ? v : 0.f;
  }
}

// ---- out = normrelu(A1@W1 + A2@W2 + bias), fused row L2 norm ---------------
template <int K1, int K2, int N>
__global__ void __launch_bounds__((N / 16) * 32)
dual_gemm_norm_relu(const float* __restrict__ A1, const float* __restrict__ W1,
                    const float* __restrict__ A2, const float* __restrict__ W2,
                    const float* __restrict__ bias, float* __restrict__ out) {
  __shared__ alignas(16) _Float16 W1s[K1 * N];
  __shared__ alignas(16) _Float16 W2s[K2 * N];
  __shared__ alignas(16) _Float16 A1s[16 * K1];
  __shared__ alignas(16) _Float16 A2s[16 * K2];
  __shared__ float Cs[16 * N];
  __shared__ float inv_norm[16];
  constexpr int S1 = K1 / 32, S2 = K2 / 32;
  const int tid  = threadIdx.x;
  const int nthr = (N / 16) * 32;
  const long long row0 = (long long)blockIdx.x * 16;

  fill_w_swz<K1, N>(W1s, W1, tid, nthr);
  fill_w_swz<K2, N>(W2s, W2, tid, nthr);
  fill_a_tile<K1>(A1s, A1, row0, tid, nthr);
  fill_a_tile<K2>(A2s, A2, row0, tid, nthr);
  __syncthreads();

  const int wave = tid >> 5, lane = tid & 31;
  v8f acc = {};
#pragma unroll
  for (int ks = 0; ks < S1; ++ks)
    acc = WMMA_F32_F16(frag_a(A1s, lane, K1, ks), frag_b(W1s, lane, S1, wave, ks), acc);
#pragma unroll
  for (int ks = 0; ks < S2; ++ks)
    acc = WMMA_F32_F16(frag_a(A2s, lane, K2, ks), frag_b(W2s, lane, S2, wave, ks), acc);

  const int col = wave * 16 + (lane & 15);
  const float bv = bias[col];
#pragma unroll
  for (int r = 0; r < 8; ++r) {
    const int rr = r + ((lane >> 4) << 3);
    Cs[rr * N + col] = acc[r] + bv;
  }
  __syncthreads();

  if (tid < 16) {
    float s = 0.f;
    for (int c = 0; c < N; ++c) { float v = Cs[tid * N + c]; s += v * v; }
    inv_norm[tid] = 1.0f / fmaxf(sqrtf(s), 1e-12f);
  }
  __syncthreads();

  for (int i = tid; i < 16 * N; i += nthr) {
    int r = i / N, c = i % N;
    float v = Cs[i] * inv_norm[r];
    out[(row0 + r) * N + c] = v > 0.f ? v : 0.f;
  }
}

// ---- agg[dst[e]][f] = max over edges of m[src[e]][f]  (m >= 0, 0-init) -----
// blockDim = 256; edge index is wave-uniform -> scalar loads for src/dst.
template <int LOGF>
__global__ void __launch_bounds__(256)
scatter_max(const float* __restrict__ m, const int* __restrict__ src,
            const int* __restrict__ dst, float* __restrict__ agg, int E) {
  const int tid = threadIdx.x;
  const int eInBlk = __builtin_amdgcn_readfirstlane(tid >> LOGF);  // wave-uniform
  const long long e = ((long long)blockIdx.x << (8 - LOGF)) + eInBlk;
  if (e >= E) return;
  const int f = tid & ((1 << LOGF) - 1);
  const int s = src[e];   // scalar (SGPR-indexed) load
  const int d = dst[e];
  const float v = m[((long long)s << LOGF) + f];
  // post-ReLU values are non-negative: int bit pattern order == float order
  atomicMax((int*)agg + (((long long)d << LOGF) + f), __float_as_int(v));
}

__global__ void zero_f32v4(float4* __restrict__ p, long long n4) {
  const long long gid = (long long)blockIdx.x * blockDim.x + threadIdx.x;
  if (gid < n4) p[gid] = make_float4(0.f, 0.f, 0.f, 0.f);
}

// ---- score = h2[src] @ W_u + h2[dst] @ W_v + b -----------------------------
__global__ void edge_score(const float* __restrict__ h2, const int* __restrict__ src,
                           const int* __restrict__ dst, const float* __restrict__ predW,
                           const float* __restrict__ predb, float* __restrict__ out, int E) {
  __shared__ float Wl[32 * 5];
  __shared__ float bl[5];
  const int tid = threadIdx.x;
  if (tid < 160) Wl[tid] = predW[tid];
  if (tid < 5)   bl[tid] = predb[tid];
  __syncthreads();
  const int e = blockIdx.x * blockDim.x + tid;
  if (e >= E) return;
  const float4* hu4 = (const float4*)(h2 + (long long)src[e] * 16);
  const float4* hv4 = (const float4*)(h2 + (long long)dst[e] * 16);
  float s[5];
#pragma unroll
  for (int c = 0; c < 5; ++c) s[c] = bl[c];
#pragma unroll
  for (int q = 0; q < 4; ++q) {
    const float4 u = hu4[q];
    const float4 v = hv4[q];
    const float uu[4] = {u.x, u.y, u.z, u.w};
    const float vv[4] = {v.x, v.y, v.z, v.w};
#pragma unroll
    for (int j = 0; j < 4; ++j) {
      const int k = q * 4 + j;
#pragma unroll
      for (int c = 0; c < 5; ++c)
        s[c] = fmaf(uu[j], Wl[k * 5 + c], fmaf(vv[j], Wl[(16 + k) * 5 + c], s[c]));
    }
  }
#pragma unroll
  for (int c = 0; c < 5; ++c) out[(long long)e * 5 + c] = s[c];
}

extern "C" void kernel_launch(void* const* d_in, const int* in_sizes, int n_in,
                              void* d_out, int out_size, void* d_ws, size_t ws_size,
                              hipStream_t stream) {
  (void)n_in; (void)out_size; (void)ws_size;
  const float* x       = (const float*)d_in[0];
  const int*   src     = (const int*)  d_in[1];
  const int*   dst     = (const int*)  d_in[2];
  const float* pool1_W = (const float*)d_in[3];
  const float* pool1_b = (const float*)d_in[4];
  const float* self1_W = (const float*)d_in[5];
  const float* neigh1_W= (const float*)d_in[6];
  const float* bias1   = (const float*)d_in[7];
  const float* pool2_W = (const float*)d_in[8];
  const float* pool2_b = (const float*)d_in[9];
  const float* self2_W = (const float*)d_in[10];
  const float* neigh2_W= (const float*)d_in[11];
  const float* bias2   = (const float*)d_in[12];
  const float* pred_W  = (const float*)d_in[13];
  const float* pred_b  = (const float*)d_in[14];
  float* out = (float*)d_out;

  const int M  = in_sizes[0] / 128;  // 100000 (divisible by 16)
  const int E  = in_sizes[1];        // 1600000
  const int MT = M / 16;             // M-tiles

  // Workspace layout (floats), 128 MB total with lifetime-based reuse:
  float* ws   = (float*)d_ws;
  float* m1   = ws;                          // [M*128]  live k1..k2
  float* agg1 = ws + (size_t)M * 128;        // [M*128]  live k2..k3
  float* h1   = ws + (size_t)M * 256;        // [M*64]   live k3..k6
  float* m2   = ws;                          // [M*64]   reuse m1 region
  float* agg2 = ws + (size_t)M * 64;         // [M*64]   reuse m1 region
  float* h2   = agg1;                        // [M*16]   reuse agg1 region

  // 1) m1 = relu(x @ pool1_W + pool1_b)                [WMMA, K=128, N=128]
  gemm_bias_relu<128, 128><<<MT, 256, 0, stream>>>(x, pool1_W, pool1_b, m1);

  // 2) agg1 = segment_max(m1[src], dst) with neg-inf -> 0 (0-init + atomicMax)
  {
    const long long n4 = (long long)M * 128 / 4;
    zero_f32v4<<<(int)((n4 + 255) / 256), 256, 0, stream>>>((float4*)agg1, n4);
    const long long blocks = ((long long)E * 128 + 255) / 256;
    scatter_max<7><<<(int)blocks, 256, 0, stream>>>(m1, src, dst, agg1, E);
  }

  // 3) h1 = normrelu(x @ self1_W + agg1 @ neigh1_W + bias1)   [WMMA, N=64]
  dual_gemm_norm_relu<128, 128, 64><<<MT, 128, 0, stream>>>(x, self1_W, agg1, neigh1_W, bias1, h1);

  // 4) m2 = relu(h1 @ pool2_W + pool2_b)               [WMMA, K=64, N=64]
  gemm_bias_relu<64, 64><<<MT, 128, 0, stream>>>(h1, pool2_W, pool2_b, m2);

  // 5) agg2 = segment_max(m2[src], dst)
  {
    const long long n4 = (long long)M * 64 / 4;
    zero_f32v4<<<(int)((n4 + 255) / 256), 256, 0, stream>>>((float4*)agg2, n4);
    const long long blocks = ((long long)E * 64 + 255) / 256;
    scatter_max<6><<<(int)blocks, 256, 0, stream>>>(m2, src, dst, agg2, E);
  }

  // 6) h2 = normrelu(h1 @ self2_W + agg2 @ neigh2_W + bias2)  [WMMA, N=16]
  dual_gemm_norm_relu<64, 64, 16><<<MT, 32, 0, stream>>>(h1, self2_W, agg2, neigh2_W, bias2, h2);

  // 7) score = h2[src] @ W_u + h2[dst] @ W_v + pred_b
  edge_score<<<(E + 255) / 256, 256, 0, stream>>>(h2, src, dst, pred_W, pred_b, out, E);
}
